// LabelAttention_38062000177791
// MI455X (gfx1250) — matrix operand order repaired and checked by reading, not compiled, not run on hardware
//
#include <hip/hip_runtime.h>
#include <hip/hip_bf16.h>
#include <math.h>

#define BB    4
#define DOUT  512
#define SEQ   4000
#define NL    8929
#define NLP   8960      // 280 tiles * 32 labels
#define ED    300
#define EDP   320       // padded K for Q-GEMM
#define LTILES 280

typedef __attribute__((ext_vector_type(16))) __bf16 v16bf;
typedef __attribute__((ext_vector_type(8)))  float  v8f;

union ABPack { uint4 u[2]; v16bf v; };
union Pack8  { uint4 u; __bf16 h[8]; };

static __device__ __forceinline__ v8f wmma_bf16(v16bf a, v16bf b, v8f c) {
  // D = A(16x32 bf16) * B(32x16 bf16) + C(16x16 f32)
  return __builtin_amdgcn_wmma_f32_16x16x32_bf16(false, a, false, b, (short)0, c, false, false);
}

// ---------------------------------------------------------------------------
// Kernel 1: pad + fp32->bf16 convert label_emb -> Lp[NLP][EDP], W_map -> Wp[DOUT][EDP]
// ---------------------------------------------------------------------------
__global__ void k_pad(const float* __restrict__ label_emb, const float* __restrict__ W_map,
                      __bf16* __restrict__ Lp, __bf16* __restrict__ Wp) {
  int i = blockIdx.x * blockDim.x + threadIdx.x;
  const int nL = NLP * EDP;
  if (i < nL) {
    int l = i / EDP, e = i % EDP;
    float v = (l < NL && e < ED) ? label_emb[l * ED + e] : 0.0f;
    Lp[i] = (__bf16)v;
  } else {
    int j = i - nL;
    if (j < DOUT * EDP) {
      int d = j / EDP, e = j % EDP;
      float v = (e < ED) ? W_map[d * ED + e] : 0.0f;
      Wp[j] = (__bf16)v;
    }
  }
}

// ---------------------------------------------------------------------------
// Kernel 2: H[b][d][s] fp32 -> HT[b][s][d] bf16 (transposed, for GEMM1 B-operand)
//                           -> Hb[b][d][s] bf16 (same layout, for GEMM2 B-operand)
// ---------------------------------------------------------------------------
__global__ void __launch_bounds__(256) k_ht(const float* __restrict__ H,
                                            __bf16* __restrict__ HT,
                                            __bf16* __restrict__ Hb) {
  __shared__ float tile[32][33];
  int s0 = blockIdx.x * 32, d0 = blockIdx.y * 32, bz = blockIdx.z;
  int si = threadIdx.x & 31, dj = threadIdx.x >> 5;   // dj 0..7
#pragma unroll
  for (int p = 0; p < 4; ++p) {
    size_t idx = ((size_t)bz * DOUT + d0 + dj + 8 * p) * SEQ + s0 + si;
    float v = H[idx];
    tile[dj + 8 * p][si] = v;
    Hb[idx] = (__bf16)v;                      // coalesced bf16 copy
  }
  __syncthreads();
  int dc = threadIdx.x & 31, sr = threadIdx.x >> 5;
#pragma unroll
  for (int p = 0; p < 4; ++p)
    HT[((size_t)bz * SEQ + s0 + sr + 8 * p) * DOUT + d0 + dc] = (__bf16)tile[dc][sr + 8 * p];
}

// ---------------------------------------------------------------------------
// Kernel 3: Qb[NLP][DOUT] bf16 = Lp @ Wp^T + b_map, via WMMA bf16, K=EDP
// ---------------------------------------------------------------------------
__global__ void __launch_bounds__(256) k_qgemm(const __bf16* __restrict__ Lp,
                                               const __bf16* __restrict__ Wp,
                                               const float* __restrict__ b_map,
                                               __bf16* __restrict__ Qb) {
  int l0 = blockIdx.x * 16;
  int wave = threadIdx.x >> 5, lane = threadIdx.x & 31;
  int lm = lane & 15; bool hi = lane >= 16;
  const __bf16* aRow = Lp + (size_t)(l0 + lm) * EDP;
  for (int dt = wave; dt < 32; dt += 8) {
    int dcol = dt * 16 + lm;
    const __bf16* bRow = Wp + (size_t)dcol * EDP;
    v8f acc = {};
#pragma unroll
    for (int k0 = 0; k0 < EDP; k0 += 32) {
      ABPack a, b;
      const uint4* ap = (const uint4*)(aRow + k0 + (hi ? 8 : 0));
      a.u[0] = ap[0]; a.u[1] = ap[2];
      const uint4* bp = (const uint4*)(bRow + k0 + (hi ? 8 : 0));
      b.u[0] = bp[0]; b.u[1] = bp[2];
      acc = wmma_bf16(a.v, b.v, acc);
    }
    float bias = b_map[dcol];
#pragma unroll
    for (int r = 0; r < 8; ++r) {
      int m = hi ? r + 8 : r;
      Qb[(size_t)(l0 + m) * DOUT + dcol] = (__bf16)(acc[r] + bias);
    }
  }
}

// ---------------------------------------------------------------------------
// Kernel 4: fused E = Q@H, softmax(seq), A out, C = A@H^T.
// Block = 32 labels x 1 batch. LDS holds E/A tile 32x4000 bf16 (250 KB).
// ---------------------------------------------------------------------------
__global__ void __launch_bounds__(256) k_main(const __bf16* __restrict__ Qb,
                                              const __bf16* __restrict__ HT,
                                              const __bf16* __restrict__ Hb,
                                              float* __restrict__ Cout,
                                              float* __restrict__ Aout) {
  extern __shared__ __align__(16) char smem[];
  __bf16* Elds = (__bf16*)smem;                 // [32][SEQ]
  const int bz = blockIdx.y;
  const int l0 = blockIdx.x * 32;
  const int tid = threadIdx.x, wave = tid >> 5, lane = tid & 31;
  const int lm = lane & 15; const bool hi = lane >= 16;
  const __bf16* HTb = HT + (size_t)bz * SEQ * DOUT;
  const __bf16* Hbb = Hb + (size_t)bz * DOUT * SEQ;

  // ---- Phase A: E tile (32 labels x 4000 seq), K = 512 dims --------------
  for (int u = wave; u < 250; u += 8) {
    int lt = u & 1, sp = u >> 1;
    int s0 = sp * 32;
    v8f acc0 = {}, acc1 = {};
    const __bf16* aRow  = Qb  + (size_t)(l0 + lt * 16 + lm) * DOUT;
    const __bf16* bRow0 = HTb + (size_t)(s0 + lm) * DOUT;
    const __bf16* bRow1 = HTb + (size_t)(s0 + 16 + lm) * DOUT;
#pragma unroll 4
    for (int k0 = 0; k0 < DOUT; k0 += 32) {
      ABPack a, b0, b1;
      const uint4* ap = (const uint4*)(aRow + k0 + (hi ? 8 : 0));
      a.u[0] = ap[0]; a.u[1] = ap[2];
      int kb2 = k0 + (hi ? 8 : 0);
      const uint4* bp0 = (const uint4*)(bRow0 + kb2);
      b0.u[0] = bp0[0]; b0.u[1] = bp0[2];
      const uint4* bp1 = (const uint4*)(bRow1 + kb2);
      b1.u[0] = bp1[0]; b1.u[1] = bp1[2];
      __builtin_prefetch(bRow0 + kb2 + 128, 0, 0);
      acc0 = wmma_bf16(a.v, b0.v, acc0);
      acc1 = wmma_bf16(a.v, b1.v, acc1);
    }
#pragma unroll
    for (int r = 0; r < 8; ++r) {
      int m = hi ? r + 8 : r;
      int row = lt * 16 + m;
      Elds[row * SEQ + s0 + lm]      = (__bf16)acc0[r];
      Elds[row * SEQ + s0 + 16 + lm] = (__bf16)acc1[r];
    }
  }
  __syncthreads();

  // ---- Phase B: softmax over seq, write A (fp32), keep probs bf16 in LDS -
  {
    int r = tid >> 3, sub = tid & 7;          // 8 lanes per row, 32 rows
    int label = l0 + r;
    uint4* erow4 = (uint4*)(Elds + r * SEQ);
    float m = -3.0e38f;
    for (int c = sub; c < 500; c += 8) {
      Pack8 p; p.u = erow4[c];
#pragma unroll
      for (int i = 0; i < 8; ++i) m = fmaxf(m, (float)p.h[i]);
    }
    for (int o = 1; o < 8; o <<= 1) m = fmaxf(m, __shfl_xor(m, o, 32));
    float s = 0.0f;
    for (int c = sub; c < 500; c += 8) {
      Pack8 p; p.u = erow4[c];
#pragma unroll
      for (int i = 0; i < 8; ++i) s += __expf((float)p.h[i] - m);
    }
    for (int o = 1; o < 8; o <<= 1) s += __shfl_xor(s, o, 32);
    float inv = 1.0f / s;
    float* arow = (label < NL) ? (Aout + ((size_t)bz * NL + label) * SEQ) : nullptr;
    for (int c = sub; c < 500; c += 8) {
      Pack8 p; p.u = erow4[c];
      float pv[8]; Pack8 q;
#pragma unroll
      for (int i = 0; i < 8; ++i) {
        pv[i] = __expf((float)p.h[i] - m) * inv;
        q.h[i] = (__bf16)pv[i];
      }
      erow4[c] = q.u;                          // probs (bf16) for GEMM2
      if (arow) {
        float4* dst = (float4*)(arow + c * 8);
        dst[0] = make_float4(pv[0], pv[1], pv[2], pv[3]);
        dst[1] = make_float4(pv[4], pv[5], pv[6], pv[7]);
      }
    }
  }
  __syncthreads();

  // ---- Phase C: C tile = A(32x4000) @ H^T(4000x512), K = seq -------------
  // 2 d-tiles per unit share one A-operand load (halves LDS traffic).
  for (int u = wave; u < 32; u += 8) {
    int lt = u & 1, dtp = u >> 1;              // dtp 0..15 -> 32 dims each
    int dcol0 = dtp * 32 + lm;
    int dcol1 = dcol0 + 16;
    v8f acc0 = {}, acc1 = {};
    const __bf16* aRow  = Elds + (lt * 16 + lm) * SEQ;
    const __bf16* hRow0 = Hbb + (size_t)dcol0 * SEQ;
    const __bf16* hRow1 = Hbb + (size_t)dcol1 * SEQ;
#pragma unroll 2
    for (int s0 = 0; s0 < SEQ; s0 += 32) {
      ABPack a, b0, b1;
      const uint4* ap = (const uint4*)(aRow + s0 + (hi ? 8 : 0));
      a.u[0] = ap[0]; a.u[1] = ap[2];
      int sb = s0 + (hi ? 8 : 0);
      const uint4* bp0 = (const uint4*)(hRow0 + sb);
      b0.u[0] = bp0[0]; b0.u[1] = bp0[2];
      const uint4* bp1 = (const uint4*)(hRow1 + sb);
      b1.u[0] = bp1[0]; b1.u[1] = bp1[2];
      __builtin_prefetch(hRow0 + sb + 256, 0, 0);
      acc0 = wmma_bf16(a.v, b0.v, acc0);
      acc1 = wmma_bf16(a.v, b1.v, acc1);
    }
#pragma unroll
    for (int r = 0; r < 8; ++r) {
      int m = hi ? r + 8 : r;
      int label = l0 + lt * 16 + m;
      if (label < NL) {
        float* crow = Cout + ((size_t)bz * NL + label) * DOUT;
        crow[dcol0] = acc0[r];
        crow[dcol1] = acc1[r];
      }
    }
  }
}

// ---------------------------------------------------------------------------
extern "C" void kernel_launch(void* const* d_in, const int* in_sizes, int n_in,
                              void* d_out, int out_size, void* d_ws, size_t ws_size,
                              hipStream_t stream) {
  (void)in_sizes; (void)n_in; (void)out_size; (void)ws_size;
  const float* H         = (const float*)d_in[0];
  const float* label_emb = (const float*)d_in[1];
  const float* W_map     = (const float*)d_in[2];
  const float* b_map     = (const float*)d_in[3];
  float* Cout = (float*)d_out;
  float* Aout = Cout + (size_t)BB * NL * DOUT;

  char* ws = (char*)d_ws;
  size_t off = 0;
  auto alloc = [&](size_t bytes) { char* p = ws + off; off += (bytes + 255) & ~(size_t)255; return p; };
  __bf16* Qb = (__bf16*)alloc((size_t)NLP * DOUT * 2);      // ~9.2 MB
  __bf16* HT = (__bf16*)alloc((size_t)BB * SEQ * DOUT * 2); // ~16.4 MB
  __bf16* Hb = (__bf16*)alloc((size_t)BB * SEQ * DOUT * 2); // ~16.4 MB
  __bf16* Lp = (__bf16*)alloc((size_t)NLP * EDP * 2);       // ~5.7 MB
  __bf16* Wp = (__bf16*)alloc((size_t)DOUT * EDP * 2);      // ~0.3 MB

  int totalPad = NLP * EDP + DOUT * EDP;
  k_pad<<<(totalPad + 255) / 256, 256, 0, stream>>>(label_emb, W_map, Lp, Wp);
  k_ht<<<dim3(SEQ / 32, DOUT / 32, BB), 256, 0, stream>>>(H, HT, Hb);
  k_qgemm<<<NLP / 16, 256, 0, stream>>>(Lp, Wp, b_map, Qb);

  const int SMEM_MAIN = 32 * SEQ * 2;                       // 250 KB LDS
  hipFuncSetAttribute((const void*)k_main, hipFuncAttributeMaxDynamicSharedMemorySize, SMEM_MAIN);
  k_main<<<dim3(LTILES, BB), 256, SMEM_MAIN, stream>>>(Qb, HT, Hb, Cout, Aout);
}
